// HypergraphLearner_83150566850958
// MI455X (gfx1250) — compile-verified, matrix operand order checked by reading, not run on hardware
//
#include <hip/hip_runtime.h>
#include <hip/hip_bf16.h>
#include <math.h>

// Problem dims (compile-time, from the reference)
constexpr int CB  = 4;
constexpr int CL  = 64;
constexpr int CE  = 32;
constexpr int CD  = 128;
constexpr int CNH = 4;
constexpr int CNL = 2;
constexpr int CN  = CL * CE;     // 2048
constexpr int CQ4 = 32;
constexpr int CH  = 32;
constexpr int CND = 2;
constexpr int CDS = CD / CNH;    // 32 per head

typedef __attribute__((ext_vector_type(16))) _Float16 v16h;
typedef __attribute__((ext_vector_type(8)))  float    v8f;

#define WMMA_F16(a, b, c) \
  __builtin_amdgcn_wmma_f32_16x16x32_f16(false, (a), false, (b), (short)0, (c), false, false)

// ---------------- WMMA fragment loader from packed f16, row-major ----------------
// Serves both A (lane = row, ks split by half-wave) and B (lane = col of W^T) since
// the index pattern is identical when the source is stored transposed-contiguous.
// Pairs are contiguous and 4B-aligned -> compiler merges into b128 loads.
__device__ inline v16h frag16(const _Float16* __restrict__ X, int ld, int row0, int k0) {
  const int lane = threadIdx.x & 31;
  const _Float16* p = X + (size_t)(row0 + (lane & 15)) * ld + k0 + ((lane >> 4) << 3);
  v16h a;
#pragma unroll
  for (int v = 0; v < 8; ++v) {
    const int kk = (v < 4) ? (v * 2) : (16 + (v - 4) * 2);
    a[2 * v]     = p[kk];
    a[2 * v + 1] = p[kk + 1];
  }
  return a;
}

// A fragment from a 16x32 row-major f16 LDS tile (P in flash attention)
__device__ inline v16h frag_a_lds(const _Float16* P) {
  const int lane = threadIdx.x & 31;
  const int m = lane & 15;
  const int hb = (lane >> 4) << 3;
  v16h a;
#pragma unroll
  for (int v = 0; v < 8; ++v) {
    const int kk = hb + ((v < 4) ? (v * 2) : (16 + (v - 4) * 2));
    a[2 * v]     = P[m * 32 + kk];
    a[2 * v + 1] = P[m * 32 + kk + 1];
  }
  return a;
}

// ---------------- f32 -> f16 conversion kernels (one-time per operand) ----------------
__global__ void k_h16(const float* __restrict__ src, _Float16* __restrict__ dst, int n) {
  const int i = blockIdx.x * blockDim.x + threadIdx.x;
  if (i < n) dst[i] = (_Float16)src[i];
}

// W (K x Nc) f32 -> WT (Nc x K) f16
__global__ void k_wt16(const float* __restrict__ W, _Float16* __restrict__ WT, int K, int Nc) {
  const int i = blockIdx.x * blockDim.x + threadIdx.x;
  if (i >= K * Nc) return;
  const int k = i / Nc, n = i % Nc;
  WT[(size_t)n * K + k] = (_Float16)W[i];
}

// V (B, CN, CD) f32 -> VT (B, CD, CN) f16
__global__ void k_vt16(const float* __restrict__ V, _Float16* __restrict__ VT) {
  const int i = blockIdx.x * blockDim.x + threadIdx.x;
  if (i >= CB * CN * CD) return;
  const int b = i / (CN * CD);
  const int rem = i % (CN * CD);
  const int n = rem / CD, d = rem % CD;
  VT[((size_t)b * CD + d) * CN + n] = (_Float16)V[i];
}

// ---------------- Batched GEMM: Y[b] = X16[b] (MxK) @ WT^T (KxNc) + bias ----------------
// Wave computes a 16x32 C strip (A fragment reused across 2 WMMAs).
// ep==0: Y = acc+bias ; ep==1: Y = RES + relu(acc+bias)
__global__ __launch_bounds__(128) void k_gemm(
    const _Float16* __restrict__ X16, int ldx, long xbs,
    const _Float16* __restrict__ WT,      // Nc x K, f16 (transposed weights)
    const float* __restrict__ bias,
    float* __restrict__ Y, int ldy, long ybs, int colOff,
    const float* __restrict__ RES, long resbs,
    int M, int K, int Nc, int ep) {
  const int b    = blockIdx.z;
  const int wave = threadIdx.x >> 5;
  const int n0   = (blockIdx.y * 4 + wave) * 32;
  if (n0 >= Nc) return;
  const int row0 = blockIdx.x * 16;
  const _Float16* Xb = X16 + (long)b * xbs;
  v8f acc0 = {}, acc1 = {};
  for (int k0 = 0; k0 < K; k0 += 32) {
    if (k0 + 32 < K)
      __builtin_prefetch(Xb + (size_t)row0 * ldx + k0 + 32, 0, 3);  // global_prefetch_b8
    v16h a  = frag16(Xb, ldx, row0, k0);
    v16h w0 = frag16(WT, K, n0, k0);
    acc0 = WMMA_F16(a, w0, acc0);
    v16h w1 = frag16(WT, K, n0 + 16, k0);
    acc1 = WMMA_F16(a, w1, acc1);
  }
  const int lane = threadIdx.x & 31;
  const int rb   = (lane >> 4) * 8;
#pragma unroll
  for (int t = 0; t < 2; ++t) {
    const v8f acc  = t ? acc1 : acc0;
    const int col  = n0 + t * 16 + (lane & 15);
    const float bv = bias ? bias[col] : 0.f;
#pragma unroll
    for (int r = 0; r < 8; ++r) {
      const int row = row0 + rb + r;
      float v = acc[r] + bv;
      if (ep == 1) {
        const float rv = RES[(long)b * resbs + (size_t)row * ldy + col];
        v = rv + (v > 0.f ? v : 0.f);
      }
      Y[(long)b * ybs + (size_t)row * ldy + colOff + col] = v;
    }
  }
}

// ---------------- Flash attention: O = qh + softmax(mask(Q Kh^T / sqrt(D))) Vh ----------------
// f16 operands (Q/K row-major, V pre-transposed); f32 Q only for the qh+ epilogue.
// maskMode 0: allowed iff xym[k] != 0 ; maskMode 1: allowed iff idx[k]==qrow && xym[k] != 0
__global__ __launch_bounds__(32) void k_flash(
    const _Float16* __restrict__ Q16, const _Float16* __restrict__ K16,
    const _Float16* __restrict__ VT, const float* __restrict__ Q32,
    float* __restrict__ Ob,
    int nq, const int* __restrict__ idx, const float* __restrict__ xym, int maskMode) {
  __shared__ _Float16 Pl[16 * 32];
  const int qt = blockIdx.x, h = blockIdx.y, b = blockIdx.z;
  const int lane = threadIdx.x & 31;
  const int nloc = lane & 15;
  const int hi   = lane >> 4;
  const int hoff = h * CDS;
  const int row0 = qt * 16;
  const _Float16* Qb  = Q16 + (size_t)b * nq * CD;
  const _Float16* Kb  = K16 + (size_t)b * CN * CD;
  const _Float16* VTb = VT + (size_t)b * CD * CN;
  const float* xy = xym + (size_t)b * CN;
  const int*   id = idx ? (idx + (size_t)b * CN) : nullptr;

  const v16h qf = frag16(Qb, CD, row0, hoff);
  v8f o0 = {}, o1 = {};
  float mi[8], li[8];
#pragma unroll
  for (int r = 0; r < 8; ++r) { mi[r] = -INFINITY; li[r] = 0.f; }
  const float scale = 0.08838834764831845f;  // 1/sqrt(128)

  for (int kb = 0; kb < CN; kb += 32) {
    // K^T B-fragments: lane owns key column kb+nloc; contiguous over head dim
    v16h bk0 = frag16(Kb, CD, kb, hoff);
    v16h bk1 = frag16(Kb, CD, kb + 16, hoff);
    v8f s0 = {}, s1 = {};
    s0 = WMMA_F16(qf, bk0, s0);
    s1 = WMMA_F16(qf, bk1, s1);

    const int k0g = kb + nloc, k1g = k0g + 16;
    const float x0 = xy[k0g], x1 = xy[k1g];
    int i0 = 0, i1 = 0;
    if (maskMode) { i0 = id[k0g]; i1 = id[k1g]; }

#pragma unroll
    for (int r = 0; r < 8; ++r) {
      const int qrow = row0 + hi * 8 + r;
      const bool ok0 = (x0 != 0.f) && (!maskMode || (i0 == qrow));
      const bool ok1 = (x1 != 0.f) && (!maskMode || (i1 == qrow));
      const float sv0 = ok0 ? s0[r] * scale : -1e9f;
      const float sv1 = ok1 ? s1[r] * scale : -1e9f;
      float m = fmaxf(sv0, sv1);
#pragma unroll
      for (int off = 1; off < 16; off <<= 1) m = fmaxf(m, __shfl_xor(m, off, 32));
      const float nm    = fmaxf(mi[r], m);
      const float alpha = __expf(mi[r] - nm);
      const float p0 = __expf(sv0 - nm);
      const float p1 = __expf(sv1 - nm);
      float rs = p0 + p1;
#pragma unroll
      for (int off = 1; off < 16; off <<= 1) rs += __shfl_xor(rs, off, 32);
      li[r] = li[r] * alpha + rs;
      mi[r] = nm;
      o0[r] *= alpha;
      o1[r] *= alpha;
      Pl[(hi * 8 + r) * 32 + nloc]      = (_Float16)p0;
      Pl[(hi * 8 + r) * 32 + 16 + nloc] = (_Float16)p1;
    }
    __syncthreads();
    v16h pf = frag_a_lds(Pl);
    // V B-fragments from transposed f16 V: contiguous over keys
    v16h v0 = frag16(VTb, CN, hoff, kb);
    v16h v1 = frag16(VTb, CN, hoff + 16, kb);
    o0 = WMMA_F16(pf, v0, o0);
    o1 = WMMA_F16(pf, v1, o1);
    __syncthreads();
  }

#pragma unroll
  for (int r = 0; r < 8; ++r) {
    const int row = row0 + hi * 8 + r;
    const float inv = 1.f / li[r];
    const size_t qb = (size_t)row * CD + hoff;
    const size_t ob = (size_t)b * nq * CD + qb;
    Ob[ob + nloc]      = Q32[(size_t)b * nq * CD + qb + nloc]      + o0[r] * inv;
    Ob[ob + 16 + nloc] = Q32[(size_t)b * nq * CD + qb + 16 + nloc] + o1[r] * inv;
  }
}

// ---------------- Spike MLP: one wave per node (H = Q4 = 32 = one lane per unit) -------------
__global__ __launch_bounds__(32) void k_spike(
    const float* __restrict__ val, const float* __restrict__ tn,
    const float* __restrict__ xym, const int* __restrict__ vidx,
    const float* __restrict__ W1, const float* __restrict__ b1,
    const float* __restrict__ W2, const float* __restrict__ b2,
    const float* __restrict__ Wg, const float* __restrict__ bg,
    const float* __restrict__ We, const float* __restrict__ be,
    const float* __restrict__ vemb,
    float* __restrict__ g_n, float* __restrict__ e_n) {
  const int node = blockIdx.x;             // b*N + n
  const int lane = threadIdx.x;
  const float mask = xym[node];
  const float f0 = val[node] * mask;
  const float f1 = tn[node];
  const float f2 = mask;
  const int vid = vidx[node];
  const float vel = vemb[vid * CH + lane];

  float acc = b1[lane] + f0 * W1[0 * CH + lane] + f1 * W1[1 * CH + lane] + f2 * W1[2 * CH + lane];
  for (int j = 0; j < CH; ++j) acc += __shfl(vel, j, 32) * W1[(3 + j) * CH + lane];
  const float h1 = 0.5f * acc * (1.0f + erff(acc * 0.7071067811865476f));  // exact GELU

  float acc2 = b2[lane];
  for (int j = 0; j < CH; ++j) acc2 += __shfl(h1, j, 32) * W2[j * CH + lane];

  float gp = acc2 * Wg[lane];
#pragma unroll
  for (int off = 16; off; off >>= 1) gp += __shfl_xor(gp, off, 32);
  const float graw = 1.0f / (1.0f + __expf(-(gp + bg[0])));
  const float gn = graw * mask;

  float ea = be[lane];
  for (int j = 0; j < CH; ++j) ea += __shfl(acc2, j, 32) * We[j * CQ4 + lane];
  e_n[(size_t)node * CQ4 + lane] = ea * gn * mask;
  if (lane == 0) g_n[node] = gn;
}

// ---------------- Elementwise helpers ----------------
__global__ void k_gate_obs(const float* __restrict__ obs, const float* __restrict__ g_n,
                           const float* __restrict__ xym, const float* __restrict__ gs,
                           float* __restrict__ out) {
  const int i = blockIdx.x * blockDim.x + threadIdx.x;
  if (i >= CB * CN * CD) return;
  const int node = i / CD;
  const float m2 = xym[node];
  const float g  = m2 + gs[0] * (g_n[node] - m2);
  out[i] = obs[i] * g;
}

__global__ void k_gather(const float* __restrict__ src, int srcRows,
                         const int* __restrict__ idx, float* __restrict__ dst) {
  const int i = blockIdx.x * blockDim.x + threadIdx.x;
  if (i >= CB * CN * CD) return;
  const int node = i / CD, d = i % CD;
  const int b = node / CN;
  dst[i] = src[((size_t)b * srcRows + idx[node]) * CD + d];
}

__global__ void k_putcols(const float* __restrict__ src, int scols,
                          float* __restrict__ dst, int dcols, int coff, int rows) {
  const int i = blockIdx.x * blockDim.x + threadIdx.x;
  if (i >= rows * scols) return;
  const int r = i / scols, c = i % scols;
  dst[(size_t)r * dcols + coff + c] = src[i];
}

// Build quaternion weight directly in transposed-f16 GEMM layout:
// WT16[out*K + in] = Wquat[out][in]
__global__ void k_quatw(const float* __restrict__ qr, const float* __restrict__ qi,
                        const float* __restrict__ qj, const float* __restrict__ qk,
                        _Float16* __restrict__ WT) {
  const int t = blockIdx.x * blockDim.x + threadIdx.x;
  if (t >= CD * CD) return;
  const int ro = t / CD, ci = t % CD;       // ro = out, ci = in
  const int p = ro >> 5, a = ro & 31, q = ci >> 5, c = ci & 31;
  const float* mats[4][4] = { { qr, qi, qj, qk },
                              { qi, qr, qk, qj },
                              { qj, qk, qr, qi },
                              { qk, qj, qi, qr } };
  const float signs[4][4] = { { 1.f, -1.f, -1.f, -1.f },
                              { 1.f,  1.f, -1.f,  1.f },
                              { 1.f,  1.f,  1.f, -1.f },
                              { 1.f, -1.f,  1.f,  1.f } };
  WT[(size_t)ro * CD + ci] = (_Float16)(signs[p][q] * mats[p][q][a * 32 + c]);
}

// masked_scatter emulation: global rank over flattened grid mask, then transpose into fine
__global__ __launch_bounds__(1024) void k_fine(const float* __restrict__ gmask,
                                               const float* __restrict__ obs,
                                               float* __restrict__ fine) {
  __shared__ int ps[1024];
  const int t = threadIdx.x;
  int flg[8]; int s = 0;
#pragma unroll
  for (int j = 0; j < 8; ++j) { flg[j] = gmask[t * 8 + j] > 0.f ? 1 : 0; s += flg[j]; }
  ps[t] = s; __syncthreads();
  for (int off = 1; off < 1024; off <<= 1) {
    int v = ps[t];
    int u = (t >= off) ? ps[t - off] : 0;
    __syncthreads();
    ps[t] = v + u;
    __syncthreads();
  }
  int run = ps[t] - s;  // exclusive prefix
  for (int j = 0; j < 8; ++j) {
    const int f = t * 8 + j;
    const int b = f / (CL * CE);
    const int rem = f % (CL * CE);
    const int l = rem / CE, e = rem % CE;
    const int rank = run;
    if (flg[j]) run++;
#pragma unroll
    for (int d = 0; d < CND; ++d) {
      const float v = flg[j] ? obs[(size_t)rank * CD + d] : 0.f;
      fine[((size_t)b * CE + e) * (CL * CND) + l * CND + d] = v;
    }
  }
}

__global__ void k_counts(const float* __restrict__ gmask, float* __restrict__ cnt) {
  const int t = blockIdx.x * blockDim.x + threadIdx.x;
  if (t >= CB * CE) return;
  const int b = t / CE, e = t % CE;
  float s = 0.f;
  for (int l = 0; l < CL; ++l) s += gmask[((size_t)b * CL + l) * CE + e];
  cnt[t] = s;
}

__global__ void k_iaa_att(const float* __restrict__ q, const float* __restrict__ k,
                          const float* __restrict__ fine, const float* __restrict__ cnt,
                          const float* __restrict__ thr, float* __restrict__ att) {
  const int t = blockIdx.x * blockDim.x + threadIdx.x;
  if (t >= CB * CE * CE) return;
  const int b = t / (CE * CE);
  const int qi = (t / CE) % CE, ki = t % CE;
  const float sc = 0.08838834764831845f;  // 1/sqrt(128) for both att and aux (dim 128)
  float a = 0.f, ax = 0.f;
  const float* qp = q + ((size_t)b * CE + qi) * CD;
  const float* kp = k + ((size_t)b * CE + ki) * CD;
  for (int d = 0; d < CD; ++d) a += qp[d] * kp[d];
  a *= sc;
  const float* f0 = fine + ((size_t)b * CE + qi) * (CL * CND);
  const float* f1 = fine + ((size_t)b * CE + ki) * (CL * CND);
  for (int d = 0; d < CL * CND; ++d) ax += f0[d] * f1[d];
  ax *= sc;
  const float mc = cnt[b * CE + qi] * cnt[b * CE + ki] * (1.f / (CL * CL));
  if ((a > thr[0]) && (ax != 0.f)) a = (1.f - mc) * a + mc * ax;
  att[t] = a;
}

__global__ __launch_bounds__(32) void k_softmax32(float* __restrict__ att) {
  const int row = blockIdx.x;
  const int lane = threadIdx.x;
  float v = att[row * CE + lane];
  float m = v;
#pragma unroll
  for (int off = 16; off; off >>= 1) m = fmaxf(m, __shfl_xor(m, off, 32));
  const float e = __expf(v - m);
  float s = e;
#pragma unroll
  for (int off = 16; off; off >>= 1) s += __shfl_xor(s, off, 32);
  att[row * CE + lane] = e / s;
}

__global__ void k_iaa_out(const float* __restrict__ att, const float* __restrict__ v,
                          float* __restrict__ out) {
  const int t = blockIdx.x * blockDim.x + threadIdx.x;
  if (t >= CB * CE * CD) return;
  const int b = t / (CE * CD);
  const int qi = (t / CD) % CE, d = t % CD;
  float s = 0.f;
  for (int k = 0; k < CE; ++k)
    s += att[((size_t)b * CE + qi) * CE + k] * v[((size_t)b * CE + k) * CD + d];
  out[t] = s;
}

__global__ void k_out(const float* __restrict__ obs, const float* __restrict__ te,
                      const float* __restrict__ va, float* __restrict__ out) {
  const int rowsTot = CN + CL + CE;
  const int i = blockIdx.x * blockDim.x + threadIdx.x;
  if (i >= CB * rowsTot * CD) return;
  const int b = i / (rowsTot * CD);
  const int rem = i % (rowsTot * CD);
  const int r = rem / CD, d = rem % CD;
  float v;
  if (r < CN)            v = obs[((size_t)b * CN + r) * CD + d];
  else if (r < CN + CL)  v = te[((size_t)b * CL + (r - CN)) * CD + d];
  else                   v = va[((size_t)b * CE + (r - CN - CL)) * CD + d];
  out[i] = v;
}

// ================= Host side =================
struct MH { const float *kb, *kw, *ob, *ow, *qb, *qw, *vb, *vw; };
struct LayerP {
  const float* gate_scale;
  MH n2t, n2v, self_;
  const float *pIb, *pIw, *pJb, *pJw, *pRb, *pRw;
  const float *quat_b, *quat_i, *quat_j, *quat_k, *quat_r;
  const float *b1b, *b1w, *b2b, *b2w, *evb, *evw, *gb, *gw, *vemb;
};

static inline void gemm(hipStream_t s, const _Float16* X16, int ldx, long xbs,
                        const _Float16* WT, const float* bias,
                        float* Y, int ldy, long ybs, int colOff,
                        const float* RES, long resbs,
                        int M, int K, int Nc, int ep) {
  dim3 g(M / 16, (Nc / 32 + 3) / 4, CB);
  k_gemm<<<g, dim3(128), 0, s>>>(X16, ldx, xbs, WT, bias, Y, ldy, ybs, colOff, RES, resbs, M, K, Nc, ep);
}

#define EW(kern, total, ...) kern<<<((total) + 255) / 256, 256, 0, stream>>>(__VA_ARGS__)

extern "C" void kernel_launch(void* const* d_in, const int* in_sizes, int n_in,
                              void* d_out, int out_size, void* d_ws, size_t ws_size,
                              hipStream_t stream) {
  (void)in_sizes; (void)n_in; (void)out_size; (void)ws_size;

  // -------- unpack flattened params (JAX pytree order: dict keys sorted) --------
  const float* P = (const float*)d_in[0];
  size_t cur = 0;
  auto take = [&](size_t n) { const float* p = P + cur; cur += n; return p; };
  auto takeMH = [&](int dk) {
    MH m;
    m.kb = take(CD); m.kw = take((size_t)dk * CD);
    m.ob = take(CD); m.ow = take((size_t)CD * CD);
    m.qb = take(CD); m.qw = take((size_t)CD * CD);
    m.vb = take(CD); m.vw = take((size_t)dk * CD);
    return m;
  };
  // top-level: "iaa" < "layers"
  const float* iaa_k_b = take(CD); const float* iaa_k_w = take((size_t)CD * CD);
  const float* iaa_q_b = take(CD); const float* iaa_q_w = take((size_t)CD * CD);
  const float* iaa_thr = take(1);
  const float* iaa_v_b = take(CD); const float* iaa_v_w = take((size_t)CD * CD);
  LayerP Lr[CNL];
  for (int i = 0; i < CNL; ++i) {
    LayerP& lp = Lr[i];
    lp.gate_scale = take(1);
    lp.n2t = takeMH(2 * CD);
    lp.n2v = takeMH(2 * CD);
    lp.pIb = take(CQ4); lp.pIw = take((size_t)CD * CQ4);
    lp.pJb = take(CQ4); lp.pJw = take((size_t)CD * CQ4);
    lp.pRb = take(CQ4); lp.pRw = take((size_t)CD * CQ4);
    lp.quat_b = take(CD);
    lp.quat_i = take(CQ4 * CQ4); lp.quat_j = take(CQ4 * CQ4);
    lp.quat_k = take(CQ4 * CQ4); lp.quat_r = take(CQ4 * CQ4);
    lp.self_ = takeMH(3 * CD);
    lp.b1b = take(CH); lp.b1w = take((size_t)(3 + CH) * CH);
    lp.b2b = take(CH); lp.b2w = take((size_t)CH * CH);
    lp.evb = take(CQ4); lp.evw = take((size_t)CH * CQ4);
    lp.gb = take(1);  lp.gw = take(CH);
    lp.vemb = take((size_t)CE * CH);
  }

  // -------- inputs --------
  const float* in_obs = (const float*)d_in[1];
  const float* in_te  = (const float*)d_in[2];
  const float* in_va  = (const float*)d_in[3];
  const int*   tidx   = (const int*)d_in[4];
  const int*   vidx   = (const int*)d_in[5];
  const float* xym    = (const float*)d_in[8];
  const float* gmask  = (const float*)d_in[9];
  const float* value  = (const float*)d_in[11];
  const float* tnorm  = (const float*)d_in[12];
  float* out = (float*)d_out;

  // -------- workspace layout (f32 region, then f16 region) --------
  float* W0 = (float*)d_ws;
  size_t off = 0;
  auto wsa = [&](size_t n) { float* p = W0 + off; off += n; return p; };
  float* obs  = wsa((size_t)CB * CN * CD);
  float* te   = wsa((size_t)CB * CL * CD);
  float* va   = wsa((size_t)CB * CE * CD);
  float* gn   = wsa((size_t)CB * CN);
  float* en   = wsa((size_t)CB * CN * CQ4);
  float* obsg = wsa((size_t)CB * CN * CD);
  float* g1   = wsa((size_t)CB * CN * CD);
  float* g2   = wsa((size_t)CB * CN * CD);
  float* kx   = wsa((size_t)CB * CN * 3 * CD);
  float* kbuf = wsa((size_t)CB * CN * CD);
  float* vbuf = wsa((size_t)CB * CN * CD);
  float* qbuf = wsa((size_t)CB * CN * CD);
  float* obuf = wsa((size_t)CB * CN * CD);
  float* o2   = wsa((size_t)CB * CN * CD);
  float* qin  = wsa((size_t)CB * CN * CD);
  float* fine = wsa((size_t)CB * CE * CL * CND);
  float* cnts = wsa((size_t)CB * CE);
  float* iq   = wsa((size_t)CB * CE * CD);
  float* ik   = wsa((size_t)CB * CE * CD);
  float* iv   = wsa((size_t)CB * CE * CD);
  float* iatt = wsa((size_t)CB * CE * CE);
  // f16 mirrors (allocated as float slots, cast; all 4B aligned)
  auto wsh = [&](size_t nHalves) { return (_Float16*)wsa((nHalves + 1) / 2); };
  _Float16* h_kx = wsh((size_t)CB * CN * 3 * CD);   // GEMM X mirror for concat inputs
  _Float16* h_a  = wsh((size_t)CB * CN * CD);       // shared sequential X mirror
  _Float16* h_q  = wsh((size_t)CB * CN * CD);
  _Float16* h_k  = wsh((size_t)CB * CN * CD);
  _Float16* h_vT = wsh((size_t)CB * CD * CN);
  _Float16* wt_q = wsh((size_t)CD * CD);
  _Float16* wt_k = wsh((size_t)3 * CD * CD);
  _Float16* wt_v = wsh((size_t)3 * CD * CD);
  _Float16* wt_o = wsh((size_t)CD * CD);
  _Float16* wt_p = wsh((size_t)CD * CQ4);
  _Float16* wt_m = wsh((size_t)CD * CD);            // quaternion / iaa weights

  hipMemcpyAsync(obs, in_obs, sizeof(float) * (size_t)CB * CN * CD, hipMemcpyDeviceToDevice, stream);
  hipMemcpyAsync(te,  in_te,  sizeof(float) * (size_t)CB * CL * CD, hipMemcpyDeviceToDevice, stream);
  hipMemcpyAsync(va,  in_va,  sizeof(float) * (size_t)CB * CE * CD, hipMemcpyDeviceToDevice, stream);

  // helper lambda: run one MHA block (q from Qsrc[M=nq rows], K/V from h_kx with width kw)
  auto mhab = [&](const MH& m, const float* Qsrc, int nq, int dk,
                  const int* midx, int maskMode, float* Ydst) {
    // projections (weights -> transposed f16, activations -> f16)
    EW(k_h16, CB * nq * CD, Qsrc, h_a, CB * nq * CD);
    EW(k_wt16, dk == 0 ? 0 : CD * CD, m.qw, wt_q, CD, CD);
    gemm(stream, h_a, CD, (long)nq * CD, wt_q, m.qb, qbuf, CD, (long)nq * CD, 0, nullptr, 0, nq, CD, CD, 0);
    EW(k_wt16, dk * CD, m.kw, wt_k, dk, CD);
    gemm(stream, h_kx, dk, (long)CN * dk, wt_k, m.kb, kbuf, CD, (long)CN * CD, 0, nullptr, 0, CN, dk, CD, 0);
    EW(k_wt16, dk * CD, m.vw, wt_v, dk, CD);
    gemm(stream, h_kx, dk, (long)CN * dk, wt_v, m.vb, vbuf, CD, (long)CN * CD, 0, nullptr, 0, CN, dk, CD, 0);
    // f16 mirrors for attention
    EW(k_h16, CB * nq * CD, qbuf, h_q, CB * nq * CD);
    EW(k_h16, CB * CN * CD, kbuf, h_k, CB * CN * CD);
    EW(k_vt16, CB * CN * CD, vbuf, h_vT);
    k_flash<<<dim3(nq / 16, CNH, CB), 32, 0, stream>>>(h_q, h_k, h_vT, qbuf, obuf, nq, midx, xym, maskMode);
    // out = o + relu(o @ Wo + bo)
    EW(k_h16, CB * nq * CD, obuf, h_a, CB * nq * CD);
    EW(k_wt16, CD * CD, m.ow, wt_o, CD, CD);
    gemm(stream, h_a, CD, (long)nq * CD, wt_o, m.ob, Ydst, CD, (long)nq * CD, 0, obuf, (long)nq * CD, nq, CD, CD, 1);
  };

  for (int i = 0; i < CNL; ++i) {
    const LayerP& lp = Lr[i];
    // spike MLP + gating
    k_spike<<<CB * CN, 32, 0, stream>>>(value, tnorm, xym, vidx,
        lp.b1w, lp.b1b, lp.b2w, lp.b2b, lp.gw, lp.gb, lp.evw, lp.evb, lp.vemb, gn, en);
    EW(k_gate_obs, CB * CN * CD, obs, gn, xym, lp.gate_scale, obsg);

    // pre-update gathers from OLD hyperedges
    EW(k_gather, CB * CN * CD, va, CE, vidx, g1);   // var_pn
    EW(k_gather, CB * CN * CD, te, CL, tidx, g2);   // temp_pn

    // ---- n2t: temp_he = mhab(temp_he, [obs_gated, var_pn]) ----
    EW(k_putcols, CB * CN * CD, obsg, CD, kx, 2 * CD, 0, CB * CN);
    EW(k_putcols, CB * CN * CD, g1,   CD, kx, 2 * CD, CD, CB * CN);
    EW(k_h16, CB * CN * 2 * CD, kx, h_kx, CB * CN * 2 * CD);
    mhab(lp.n2t, te, CL, 2 * CD, tidx, 1, te);

    // ---- n2v: var_he = mhab(var_he, [obs_gated, temp_pn]) ----
    EW(k_putcols, CB * CN * CD, g2, CD, kx, 2 * CD, CD, CB * CN);  // col0 still obs_gated
    EW(k_h16, CB * CN * 2 * CD, kx, h_kx, CB * CN * 2 * CD);
    mhab(lp.n2v, va, CE, 2 * CD, vidx, 1, va);

    // post-update gathers
    EW(k_gather, CB * CN * CD, te, CL, tidx, g1);   // temp_pn2
    EW(k_gather, CB * CN * CD, va, CE, vidx, g2);   // var_pn2

    // ---- self: o = mhab(obs, [obs_gated, temp_pn2, var_pn2]) ----
    EW(k_putcols, CB * CN * CD, obsg, CD, kx, 3 * CD, 0, CB * CN);
    EW(k_putcols, CB * CN * CD, g1,   CD, kx, 3 * CD, CD, CB * CN);
    EW(k_putcols, CB * CN * CD, g2,   CD, kx, 3 * CD, 2 * CD, CB * CN);
    EW(k_h16, CB * CN * 3 * CD, kx, h_kx, CB * CN * 3 * CD);
    mhab(lp.self_, obs, CN, 3 * CD, nullptr, 0, o2);

    // ---- quaternion mix: obs = quat([qR | qI | qJ | e_n]) ----
    EW(k_h16, CB * CN * CD, o2, h_a, CB * CN * CD);
    EW(k_wt16, CD * CQ4, lp.pRw, wt_p, CD, CQ4);
    gemm(stream, h_a, CD, (long)CN * CD, wt_p, lp.pRb, qin, CD, (long)CN * CD, 0, nullptr, 0, CN, CD, CQ4, 0);
    EW(k_h16, CB * CN * CD, g1, h_a, CB * CN * CD);
    EW(k_wt16, CD * CQ4, lp.pIw, wt_p, CD, CQ4);
    gemm(stream, h_a, CD, (long)CN * CD, wt_p, lp.pIb, qin, CD, (long)CN * CD, 32, nullptr, 0, CN, CD, CQ4, 0);
    EW(k_h16, CB * CN * CD, g2, h_a, CB * CN * CD);
    EW(k_wt16, CD * CQ4, lp.pJw, wt_p, CD, CQ4);
    gemm(stream, h_a, CD, (long)CN * CD, wt_p, lp.pJb, qin, CD, (long)CN * CD, 64, nullptr, 0, CN, CD, CQ4, 0);
    EW(k_putcols, CB * CN * CQ4, en, CQ4, qin, CD, 96, CB * CN);
    EW(k_quatw, CD * CD, lp.quat_r, lp.quat_i, lp.quat_j, lp.quat_k, wt_m);
    EW(k_h16, CB * CN * CD, qin, h_a, CB * CN * CD);
    gemm(stream, h_a, CD, (long)CN * CD, wt_m, lp.quat_b, obs, CD, (long)CN * CD, 0, nullptr, 0, CN, CD, CD, 0);

    // ---- last layer: fine-grained scatter + IAA on var_he ----
    if (i == CNL - 1) {
      k_fine<<<1, 1024, 0, stream>>>(gmask, obs, fine);
      EW(k_counts, CB * CE, gmask, cnts);
      EW(k_h16, CB * CE * CD, va, h_a, CB * CE * CD);
      EW(k_wt16, CD * CD, iaa_q_w, wt_m, CD, CD);
      gemm(stream, h_a, CD, (long)CE * CD, wt_m, iaa_q_b, iq, CD, (long)CE * CD, 0, nullptr, 0, CE, CD, CD, 0);
      EW(k_wt16, CD * CD, iaa_k_w, wt_m, CD, CD);
      gemm(stream, h_a, CD, (long)CE * CD, wt_m, iaa_k_b, ik, CD, (long)CE * CD, 0, nullptr, 0, CE, CD, CD, 0);
      EW(k_wt16, CD * CD, iaa_v_w, wt_m, CD, CD);
      gemm(stream, h_a, CD, (long)CE * CD, wt_m, iaa_v_b, iv, CD, (long)CE * CD, 0, nullptr, 0, CE, CD, CD, 0);
      EW(k_iaa_att, CB * CE * CE, iq, ik, fine, cnts, iaa_thr, iatt);
      k_softmax32<<<CB * CE, 32, 0, stream>>>(iatt);
      EW(k_iaa_out, CB * CE * CD, iatt, iv, va);
    }
  }

  // ---- final concat [obs ; temp ; var] ----
  EW(k_out, CB * (CN + CL + CE) * CD, obs, te, va, out);
}
